// Net_58025008169662
// MI455X (gfx1250) — matrix operand order repaired and checked by reading, not compiled, not run on hardware
//
#include <hip/hip_runtime.h>
#include <hip/hip_bf16.h>
#include <hip/hip_fp16.h>

typedef _Float16 v16h __attribute__((ext_vector_type(16)));
typedef _Float16 v8h  __attribute__((ext_vector_type(8)));
typedef _Float16 v4h  __attribute__((ext_vector_type(4)));
typedef float    v8f  __attribute__((ext_vector_type(8)));

#define N_NODES   100000
#define N_EDGES   1600000
#define IN_FEAT   512
#define HIDDEN    128
#define N_CLASSES 7

#if __has_builtin(__builtin_amdgcn_sched_barrier)
#define SCHED_FENCE() __builtin_amdgcn_sched_barrier(0)
#else
#define SCHED_FENCE()
#endif

// ---------------------------------------------------------------------------
// Degree: deg[i] = 1 (self loop) + sum_{e: dst[e]==i} w[e]; then dinv = rsqrt.
// ---------------------------------------------------------------------------
__global__ void deg_init_kernel(float* deg) {
    int i = blockIdx.x * blockDim.x + threadIdx.x;
    if (i < N_NODES) deg[i] = 1.0f;  // self-loop weight
}

__global__ void deg_acc_kernel(const int* __restrict__ dst,
                               const float* __restrict__ w,
                               float* deg) {
    int e = blockIdx.x * blockDim.x + threadIdx.x;
    if (e < N_EDGES) atomicAdd(&deg[dst[e]], w[e]);
}

__global__ void deg_rsqrt_kernel(float* deg) {
    int i = blockIdx.x * blockDim.x + threadIdx.x;
    if (i < N_NODES) deg[i] = rsqrtf(deg[i]);   // deg >= 1 always
}

// ---------------------------------------------------------------------------
// H = X @ W1  (100000x512 * 512x128), f32 -> f16 staged in LDS, WMMA f32 acc.
// Block: 256 threads = 8 waves; tile = 128 rows x 128 cols; wave owns 16 rows.
// Fully unrolled k-loop with ping-pong LDS buffers: global b128 loads for
// tile k+1 are pinned (sched_barrier) before the WMMAs of tile k; the single
// loadcnt drain happens in the convert+ds_store phase after the WMMAs.
// OOB tail rows are address-clamped only: tile row r feeds only output row
// row0+r, and the H store is predicated, so no value zeroing is needed.
// ---------------------------------------------------------------------------
__launch_bounds__(256)
__global__ void gemm_xw1_kernel(const float* __restrict__ X,
                                const float* __restrict__ W,
                                float* __restrict__ H) {
    __shared__ __align__(16) _Float16 As[2][128][40];   // 80 B/row (16B-aligned)
    __shared__ __align__(16) _Float16 Bs[2][32][136];   // 272 B/row (16B-aligned)

    const int tid  = threadIdx.x;
    const int wave = tid >> 5;
    const int lane = tid & 31;
    const int row0 = blockIdx.x * 128;

    // staging geometry:
    //  X tile 128x32 f32: thread t owns rows (t>>3)+i*32, cols (t&7)*4..+3
    //  W tile  32x128 f32: thread t owns float4 index t+i*256
    const int xr = tid >> 3;
    const int xc = (tid & 7) * 4;

    float4 xreg[4], wreg[4];

    // fragment geometry (ISA 7.12.2):
    const int m  = lane & 15;                 // A: lane -> row M ; C: N = lane&15
    const int hh = lane >> 4;                 // half-wave selector
    const int kb = hh * 8;                    // A: lanes 16-31 start at K=8
    const int kk = (lane & 15) + (hh << 4);   // B: lane -> row K

    v8f acc[8] = {};

#define LOAD_TILES(K0)                                                         \
    do {                                                                       \
        _Pragma("unroll")                                                      \
        for (int i = 0; i < 4; ++i) {                                          \
            int grow = row0 + xr + i * 32;                                     \
            int gc   = grow < N_NODES ? grow : 0; /* clamp: tail is unused */  \
            xreg[i]  = *(const float4*)&X[(size_t)gc * IN_FEAT + (K0) + xc];   \
        }                                                                      \
        _Pragma("unroll")                                                      \
        for (int i = 0; i < 4; ++i) {                                          \
            int e = tid + i * 256;                                             \
            int r = e >> 5, c = (e & 31) * 4;                                  \
            wreg[i] = *(const float4*)&W[(size_t)((K0) + r) * HIDDEN + c];     \
        }                                                                      \
    } while (0)

#define STORE_TILES(BUF)                                                       \
    do {                                                                       \
        _Pragma("unroll")                                                      \
        for (int i = 0; i < 4; ++i) {                                          \
            int r = xr + i * 32;                                               \
            float4 v = xreg[i];                                                \
            v4h p; p[0] = (_Float16)v.x; p[1] = (_Float16)v.y;                 \
                   p[2] = (_Float16)v.z; p[3] = (_Float16)v.w;                 \
            *(v4h*)&As[BUF][r][xc] = p;                                        \
        }                                                                      \
        _Pragma("unroll")                                                      \
        for (int i = 0; i < 4; ++i) {                                          \
            int e = tid + i * 256;                                             \
            int r = e >> 5, c = (e & 31) * 4;                                  \
            float4 v = wreg[i];                                                \
            v4h p; p[0] = (_Float16)v.x; p[1] = (_Float16)v.y;                 \
                   p[2] = (_Float16)v.z; p[3] = (_Float16)v.w;                 \
            *(v4h*)&Bs[BUF][r][c] = p;                                         \
        }                                                                      \
    } while (0)

    LOAD_TILES(0);
    STORE_TILES(0);
    __syncthreads();

    const int NSTEP = IN_FEAT / 32;   // 16
    #pragma unroll
    for (int step = 0; step < NSTEP; ++step) {
        const int cur = step & 1;     // compile-time constant after unroll

        // issue next tile's global loads; fence keeps them above the WMMAs
        if (step + 1 < NSTEP) {
            LOAD_TILES((step + 1) * 32);
            SCHED_FENCE();
        }

        // ---- A fragment: v16h; elem i -> K = kb + (i&7) + 16*(i>=8) --------
        v8h alo = *(const v8h*)&As[cur][wave * 16 + m][kb];
        v8h ahi = *(const v8h*)&As[cur][wave * 16 + m][kb + 16];
        v16h a  = __builtin_shufflevector(alo, ahi,
                     0,1,2,3,4,5,6,7,8,9,10,11,12,13,14,15);

        // ---- 8 column tiles of 16: B lane=N, elem i -> K = kb + (i&7) + 16*(i>=8)
        #pragma unroll
        for (int nt = 0; nt < 8; ++nt) {
            v16h b;
            #pragma unroll
            for (int s = 0; s < 16; ++s)
                b[s] = Bs[cur][kb + (s & 7) + ((s >> 3) << 4)][nt * 16 + m];
            acc[nt] = __builtin_amdgcn_wmma_f32_16x16x32_f16(
                          false, a, false, b, (short)0, acc[nt], false, false);
        }

        // convert + stage into the other buffer while this buffer is consumed
        if (step + 1 < NSTEP) STORE_TILES(cur ^ 1);
        __syncthreads();
    }
#undef LOAD_TILES
#undef STORE_TILES

    // ---- store: C layout VGPR r -> M = r + 8*(lane>=16), N = lane&15 -------
    #pragma unroll
    for (int nt = 0; nt < 8; ++nt) {
        #pragma unroll
        for (int r = 0; r < 8; ++r) {
            int grow = row0 + wave * 16 + r + hh * 8;
            if (grow < N_NODES)
                H[(size_t)grow * HIDDEN + nt * 16 + m] = acc[nt][r];
        }
    }
}

// ---------------------------------------------------------------------------
// out1[i][:] = H[i][:] * dinv[i]^2   (self-loop contribution, also inits buf)
// ---------------------------------------------------------------------------
__global__ void selfloop128_kernel(const float* __restrict__ H,
                                   const float* __restrict__ dinv,
                                   float* __restrict__ out) {
    size_t i = (size_t)blockIdx.x * blockDim.x + threadIdx.x;
    if (i < (size_t)N_NODES * HIDDEN) {
        int node = (int)(i >> 7);
        float d  = dinv[node];
        out[i] = H[i] * d * d;
    }
}

// ---------------------------------------------------------------------------
// Edge scatter (128 feats): one wave per edge, float4 per lane, L2 atomics.
// ---------------------------------------------------------------------------
__launch_bounds__(256)
__global__ void scatter128_kernel(const int* __restrict__ src,
                                  const int* __restrict__ dst,
                                  const float* __restrict__ w,
                                  const float* __restrict__ dinv,
                                  const float* __restrict__ H,
                                  float* __restrict__ out) {
    long long gt = (long long)blockIdx.x * 256 + threadIdx.x;
    int e    = (int)(gt >> 5);
    int lane = threadIdx.x & 31;
    if (e >= N_EDGES) return;
    int s = src[e], d = dst[e];
    float c = dinv[s] * w[e] * dinv[d];
    const float4 v = *(const float4*)&H[(size_t)s * HIDDEN + lane * 4];
    float* o = &out[(size_t)d * HIDDEN + lane * 4];
    atomicAdd(o + 0, v.x * c);
    atomicAdd(o + 1, v.y * c);
    atomicAdd(o + 2, v.z * c);
    atomicAdd(o + 3, v.w * c);
}

// ---------------------------------------------------------------------------
// h2 = elu(out1 + b1)   (in place)
// ---------------------------------------------------------------------------
__global__ void bias_elu_kernel(float* __restrict__ A,
                                const float* __restrict__ b1) {
    size_t i = (size_t)blockIdx.x * blockDim.x + threadIdx.x;
    if (i < (size_t)N_NODES * HIDDEN) {
        float v = A[i] + b1[i & 127];
        A[i] = (v > 0.0f) ? v : expm1f(v);
    }
}

// ---------------------------------------------------------------------------
// G = h2 @ W2 (100000x128 * 128x7): 8 threads per node (j = lane%8, j<7).
// ---------------------------------------------------------------------------
__global__ void gemm_w2_kernel(const float* __restrict__ H2,
                               const float* __restrict__ W2,
                               float* __restrict__ G) {
    long long t = (long long)blockIdx.x * blockDim.x + threadIdx.x;
    int node = (int)(t >> 3);
    int j    = (int)(t & 7);
    if (node >= N_NODES || j >= N_CLASSES) return;
    float acc = 0.0f;
    const float* hrow = &H2[(size_t)node * HIDDEN];
    for (int k = 0; k < HIDDEN; ++k)
        acc = fmaf(hrow[k], W2[k * N_CLASSES + j], acc);
    G[(size_t)node * N_CLASSES + j] = acc;
}

__global__ void selfloop7_kernel(const float* __restrict__ G,
                                 const float* __restrict__ dinv,
                                 float* __restrict__ out) {
    int i = blockIdx.x * blockDim.x + threadIdx.x;
    if (i < N_NODES * N_CLASSES) {
        int node = i / N_CLASSES;
        float d  = dinv[node];
        out[i] = G[i] * d * d;
    }
}

__global__ void scatter7_kernel(const int* __restrict__ src,
                                const int* __restrict__ dst,
                                const float* __restrict__ w,
                                const float* __restrict__ dinv,
                                const float* __restrict__ G,
                                float* __restrict__ out) {
    long long t = (long long)blockIdx.x * blockDim.x + threadIdx.x;
    int e = (int)(t >> 3);
    int j = (int)(t & 7);
    if (e >= N_EDGES || j >= N_CLASSES) return;
    int s = src[e], d = dst[e];
    float c = dinv[s] * w[e] * dinv[d];
    atomicAdd(&out[(size_t)d * N_CLASSES + j], G[(size_t)s * N_CLASSES + j] * c);
}

// ---------------------------------------------------------------------------
// out = log_softmax(out + b2) rowwise (7 classes), in place.
// ---------------------------------------------------------------------------
__global__ void logsoftmax_kernel(float* __restrict__ out,
                                  const float* __restrict__ b2) {
    int n = blockIdx.x * blockDim.x + threadIdx.x;
    if (n >= N_NODES) return;
    float v[N_CLASSES];
    float m = -1e30f;
    for (int j = 0; j < N_CLASSES; ++j) {
        v[j] = out[(size_t)n * N_CLASSES + j] + b2[j];
        m = fmaxf(m, v[j]);
    }
    float s = 0.0f;
    for (int j = 0; j < N_CLASSES; ++j) s += expf(v[j] - m);
    float ls = logf(s);
    for (int j = 0; j < N_CLASSES; ++j)
        out[(size_t)n * N_CLASSES + j] = v[j] - m - ls;
}

// ---------------------------------------------------------------------------
extern "C" void kernel_launch(void* const* d_in, const int* in_sizes, int n_in,
                              void* d_out, int out_size, void* d_ws, size_t ws_size,
                              hipStream_t stream) {
    (void)in_sizes; (void)n_in; (void)out_size; (void)ws_size;
    const float* X    = (const float*)d_in[0];
    const int*   ei   = (const int*)d_in[1];
    const float* ew   = (const float*)d_in[2];
    const float* W1   = (const float*)d_in[3];
    const float* b1   = (const float*)d_in[4];
    const float* W2   = (const float*)d_in[5];
    const float* b2   = (const float*)d_in[6];
    const int* src = ei;
    const int* dst = ei + N_EDGES;
    float* OUT = (float*)d_out;

    // workspace carve (all 16B-aligned offsets)
    char*  ws   = (char*)d_ws;
    float* H    = (float*)(ws);                    // 51,200,000 B
    float* OUT1 = (float*)(ws + 51200000);         // 51,200,000 B
    float* DINV = (float*)(ws + 102400000);        //    400,000 B
    float* G    = (float*)(ws + 102800000);        //  2,800,000 B

    const int T = 256;

    // degrees -> dinv
    deg_init_kernel <<<(N_NODES + T - 1) / T, T, 0, stream>>>(DINV);
    deg_acc_kernel  <<<(N_EDGES + T - 1) / T, T, 0, stream>>>(dst, ew, DINV);
    deg_rsqrt_kernel<<<(N_NODES + T - 1) / T, T, 0, stream>>>(DINV);

    // layer 1: transform (WMMA), self-loop init, edge scatter, bias+ELU
    gemm_xw1_kernel<<<(N_NODES + 127) / 128, T, 0, stream>>>(X, W1, H);
    selfloop128_kernel<<<(N_NODES * HIDDEN + T - 1) / T, T, 0, stream>>>(H, DINV, OUT1);
    scatter128_kernel<<<(int)(((long long)N_EDGES * 32 + T - 1) / T), T, 0, stream>>>(
        src, dst, ew, DINV, H, OUT1);
    bias_elu_kernel<<<(N_NODES * HIDDEN + T - 1) / T, T, 0, stream>>>(OUT1, b1);

    // layer 2: transform, self-loop init (into d_out), edge scatter, softmax
    gemm_w2_kernel<<<(int)(((long long)N_NODES * 8 + T - 1) / T), T, 0, stream>>>(OUT1, W2, G);
    selfloop7_kernel<<<(N_NODES * N_CLASSES + T - 1) / T, T, 0, stream>>>(G, DINV, OUT);
    scatter7_kernel<<<(int)(((long long)N_EDGES * 8 + T - 1) / T), T, 0, stream>>>(
        src, dst, ew, DINV, G, OUT);
    logsoftmax_kernel<<<(N_NODES + T - 1) / T, T, 0, stream>>>(OUT, b2);
}
